// GEN_40544491274927
// MI455X (gfx1250) — matrix operand order repaired and checked by reading, not compiled, not run on hardware
//
#include <hip/hip_runtime.h>
#include <cstdint>
#include <cstddef>

typedef __attribute__((ext_vector_type(16))) _Float16 v16h;
typedef __attribute__((ext_vector_type(8)))  float    v8f;

constexpr int CB  = 4;
constexpr int CNP = 4096;
constexpr int CNQ = 2048;
constexpr int CNN = 2048;
constexpr int CE  = 32768;
constexpr int CDH = 128;

// ---------------------------------------------------------------------------
// CDNA5 async global->LDS copy (GLOBAL_LOAD_ASYNC_TO_LDS_B128, ASYNCcnt),
// compile-safe: falls back to plain 16B copies if the builtin is absent.
// Builtin signature (from hipcc diagnostic): params are generic pointers to
// 'int __attribute__((vector_size(16)))'.
// ---------------------------------------------------------------------------
#if defined(__HIP_DEVICE_COMPILE__) && __has_builtin(__builtin_amdgcn_global_load_async_to_lds_b128)
#define HAVE_ASYNC_LDS 1
#else
#define HAVE_ASYNC_LDS 0
#endif

typedef int v4i_vs __attribute__((vector_size(4 * sizeof(int))));

__device__ inline void async_copy16(const _Float16* g, _Float16* l) {
#if HAVE_ASYNC_LDS
  __builtin_amdgcn_global_load_async_to_lds_b128(
      (v4i_vs*)const_cast<_Float16*>(g), (v4i_vs*)l, 0, 0);
#else
  *(float4*)l = *(const float4*)g;
#endif
}

__device__ inline void async_wait0() {
#if HAVE_ASYNC_LDS
#if __has_builtin(__builtin_amdgcn_s_wait_asynccnt)
  __builtin_amdgcn_s_wait_asynccnt(0);
#else
  asm volatile("s_wait_asynccnt 0x0" ::: "memory");
#endif
#endif
}

__device__ inline v8f vzero8() {
  v8f z = {0.f, 0.f, 0.f, 0.f, 0.f, 0.f, 0.f, 0.f};
  return z;
}

__device__ inline v8f wmma16(v16h a, v16h b, v8f c) {
  // D = A(16x32 f16) * B(32x16 f16) + C(16x16 f32)
  return __builtin_amdgcn_wmma_f32_16x16x32_f16(false, a, false, b, (short)0, c, false, false);
}

// A fragment: 16x32 f16 from row-major storage, leading dim lda.
// ISA layout: lanes 0-15 -> rows M=0..15 holding K 0-7,16-23; lanes 16-31 same rows, K 8-15,24-31.
__device__ inline v16h load_a_frag(const _Float16* __restrict__ base, int lda) {
  const int lane = threadIdx.x & 31;
  const int row  = lane & 15;
  const int kb   = (lane >> 4) << 3;   // 0 or 8
  const _Float16* p = base + row * lda + kb;
  v16h a;
#pragma unroll
  for (int j = 0; j < 8; ++j) a[j] = p[j];
#pragma unroll
  for (int j = 0; j < 8; ++j) a[8 + j] = p[16 + j];
  return a;
}

// B fragment: 32x16 f16 read from TRANSPOSED storage Bt[n][k] (leading dim ldb).
// lanes 0-15 hold K 0-7,16-23 for their column; lanes 16-31 hold K 8-15,24-31.
__device__ inline v16h load_bt_frag(const _Float16* __restrict__ base, int ldb) {
  const int lane = threadIdx.x & 31;
  const int col  = lane & 15;
  const int kb   = (lane >> 4) << 3;   // 0 or 8
  const _Float16* p = base + col * ldb + kb;
  v16h b;
#pragma unroll
  for (int j = 0; j < 8; ++j) b[j] = p[j];
#pragma unroll
  for (int j = 0; j < 8; ++j) b[8 + j] = p[16 + j];
  return b;
}

// Generic per-workgroup GEMM stage of one MLP layer.
// actIn: LDS f16 [rows][Kin] row-major (each wave owns a 16-row strip).
// Wt:    f16 [ntiles*16][Kin]  (W transposed, zero-padded).
// epi(row, col, value+bias) consumes the fp32 result.
template <typename Epi>
__device__ inline void mlp_gemm(const _Float16* __restrict__ actIn, int Kin,
                                const _Float16* __restrict__ Wt,
                                const float* __restrict__ bias,
                                int ntiles, Epi epi) {
  const int wave = threadIdx.x >> 5;
  const int lane = threadIdx.x & 31;
  const int row0 = wave * 16;
  const int half = lane >> 4;
  const int nsub = lane & 15;
  for (int t = 0; t < ntiles; ++t) {
    v8f acc = vzero8();
    for (int ks = 0; ks < Kin; ks += 32) {
      v16h a = load_a_frag(actIn + row0 * Kin + ks, Kin);
      v16h b = load_bt_frag(Wt + (t * 16) * Kin + ks, Kin);
      acc = wmma16(a, b, acc);
    }
    const int col = t * 16 + nsub;
    const float bv = bias ? bias[col] : 0.f;
    const int rbase = row0 + half * 8;
#pragma unroll
    for (int r = 0; r < 8; ++r) epi(rbase + r, col, acc[r] + bv);
  }
}

__device__ inline void mlp_layer_relu(const _Float16* actIn, int Kin, _Float16* actOut,
                                      const _Float16* Wt, const float* bias) {
  mlp_gemm(actIn, Kin, Wt, bias, 8, [&](int r, int c, float v) {
    actOut[r * CDH + c] = (_Float16)fmaxf(v, 0.f);
  });
  __syncthreads();
}

// ---------------------------------------------------------------------------
// Weight convert: f32 W[r][c] -> f16 Wt[nPad][kPad] (transposed, zero padded)
// ---------------------------------------------------------------------------
__global__ void k_convert(const float* __restrict__ src, _Float16* __restrict__ dst,
                          int r, int c, int kPad, int nPad) {
  int i = blockIdx.x * blockDim.x + threadIdx.x;
  if (i >= kPad * nPad) return;
  int cc = i / kPad;   // output channel (column of W)
  int rr = i % kPad;   // input channel (row of W)
  float v = (cc < c && rr < r) ? src[rr * c + cc] : 0.f;
  dst[i] = (_Float16)v;
}

// ---------------------------------------------------------------------------
// Softmax pass 1: per point, max(-d2) and 1/sum(exp) over all NN nodes.
// ---------------------------------------------------------------------------
__global__ void k_stats(const float* __restrict__ pts, const float* __restrict__ P,
                        float* __restrict__ stats, int rowsTotal) {
  __shared__ float sP[CNN * 2];
  for (int i = threadIdx.x; i < CNN * 2; i += blockDim.x) sP[i] = P[i];
  __syncthreads();
  int row = blockIdx.x * blockDim.x + threadIdx.x;
  if (row >= rowsTotal) return;
  float x0 = pts[(size_t)row * 2], x1 = pts[(size_t)row * 2 + 1];
  float m = -3.4e38f;
  for (int n = 0; n < CNN; ++n) {
    float d0 = x0 - sP[2 * n], d1 = x1 - sP[2 * n + 1];
    float nd2 = -(d0 * d0 + d1 * d1);
    m = fmaxf(m, nd2);
  }
  float Z = 0.f;
  for (int n = 0; n < CNN; ++n) {
    float d0 = x0 - sP[2 * n], d1 = x1 - sP[2 * n + 1];
    Z += __expf(-(d0 * d0 + d1 * d1) - m);
  }
  stats[(size_t)row * 2]     = m;
  stats[(size_t)row * 2 + 1] = 1.f / Z;
}

// ---------------------------------------------------------------------------
// Encoder: [x|s] (K padded 5->32) -> 128 -> 128 -> 128, fused in LDS.
// 64 rows / block, 128 threads (4 waves).
// ---------------------------------------------------------------------------
__global__ void k_encoder(const float* __restrict__ x, const float* __restrict__ s,
                          const _Float16* __restrict__ wt0, const float* __restrict__ b0,
                          const _Float16* __restrict__ wt1, const float* __restrict__ b1,
                          const _Float16* __restrict__ wt2, const float* __restrict__ b2,
                          _Float16* __restrict__ emb) {
  __shared__ __align__(16) _Float16 bufA[64 * 128];
  __shared__ __align__(16) _Float16 bufB[64 * 128];
  const int r0 = blockIdx.x * 64;
  for (int i = threadIdx.x; i < 64 * 32; i += blockDim.x) {
    int row = i >> 5, c = i & 31;
    size_t gr = (size_t)(r0 + row);
    float v = 0.f;
    if (c < 2)      v = x[gr * 2 + c];
    else if (c < 5) v = s[gr * 3 + (c - 2)];
    bufA[i] = (_Float16)v;
  }
  __syncthreads();
  mlp_layer_relu(bufA, 32,  bufB, wt0, b0);
  mlp_layer_relu(bufB, 128, bufA, wt1, b1);
  mlp_gemm(bufA, 128, wt2, b2, 8, [&](int r, int c, float v) {
    emb[(size_t)(r0 + r) * 128 + c] = (_Float16)v;
  });
}

// ---------------------------------------------------------------------------
// latents = softmax-scores^T @ emb  (per batch: 2048 x 4096 x 128).
// Score A-tiles generated on the fly (fused exp) -> no 64MB score buffer.
// 128-row tiles, 256 threads (8 waves), persistent 8x v8f accumulators.
// ---------------------------------------------------------------------------
__global__ void k_agg(const float* __restrict__ xp, const float* __restrict__ P,
                      const float* __restrict__ statsX, const _Float16* __restrict__ emb,
                      float* __restrict__ latF32, _Float16* __restrict__ latF16) {
  __shared__ __align__(16) _Float16 At[128 * 32];
  __shared__ __align__(16) _Float16 Bt[128 * 32];
  __shared__ float sPn[128 * 2];
  const int b  = blockIdx.y;
  const int n0 = blockIdx.x * 128;
  const int tid = threadIdx.x;
  const int wave = tid >> 5, lane = tid & 31;
  sPn[tid] = P[(size_t)n0 * 2 + tid];   // 256 threads = 256 floats
  v8f acc[8];
#pragma unroll
  for (int t = 0; t < 8; ++t) acc[t] = vzero8();

  for (int p0 = 0; p0 < CNP; p0 += 32) {
    __syncthreads();
    // prefetch next emb slab (one cacheline per row) while this slab computes
    if (p0 + 32 < CNP && tid < 32)
      __builtin_prefetch(emb + ((size_t)b * CNP + p0 + 32 + tid) * 128, 0, 3);
    // stage emb^T slab: Bt[d][k]
    for (int i = tid; i < 32 * 128; i += 256) {
      int k = i >> 7, d = i & 127;
      Bt[d * 32 + k] = emb[((size_t)b * CNP + p0 + k) * 128 + d];
    }
    // generate At[n][p] = exp(-d2 - m_p) * invZ_p  (fused softmax)
    for (int i = tid; i < 128 * 32; i += 256) {
      int n = i >> 5, pp = i & 31;
      size_t pg = (size_t)b * CNP + p0 + pp;
      float x0 = xp[pg * 2], x1 = xp[pg * 2 + 1];
      float m  = statsX[pg * 2], iz = statsX[pg * 2 + 1];
      float d0 = x0 - sPn[2 * n], d1 = x1 - sPn[2 * n + 1];
      At[i] = (_Float16)(__expf(-(d0 * d0 + d1 * d1) - m) * iz);
    }
    __syncthreads();
    v16h a = load_a_frag(At + (wave * 16) * 32, 32);
#pragma unroll
    for (int t = 0; t < 8; ++t) {
      v16h bf = load_bt_frag(Bt + (t * 16) * 32, 32);
      acc[t] = wmma16(a, bf, acc[t]);
    }
  }
  const int half = lane >> 4, nsub = lane & 15;
#pragma unroll
  for (int t = 0; t < 8; ++t) {
    int col = t * 16 + nsub;
#pragma unroll
    for (int r = 0; r < 8; ++r) {
      int row = wave * 16 + half * 8 + r;
      size_t off = ((size_t)b * CNN + n0 + row) * 128 + col;
      float v = acc[t][r];
      latF32[off] = v;
      latF16[off] = (_Float16)v;
    }
  }
}

// ---------------------------------------------------------------------------
// z = softmax-scores(q) @ latents  (per batch: 2048 x 2048 x 128), same fusion.
// ---------------------------------------------------------------------------
__global__ void k_zq(const float* __restrict__ qp, const float* __restrict__ P,
                     const float* __restrict__ statsQ, const _Float16* __restrict__ latF16,
                     _Float16* __restrict__ zf16) {
  __shared__ __align__(16) _Float16 At[128 * 32];
  __shared__ __align__(16) _Float16 Bt[128 * 32];
  __shared__ float sq[128 * 2];
  __shared__ float sst[128 * 2];
  const int b  = blockIdx.y;
  const int q0 = blockIdx.x * 128;
  const int tid = threadIdx.x;
  const int wave = tid >> 5, lane = tid & 31;
  sq[tid]  = qp[((size_t)b * CNQ + q0) * 2 + tid];
  sst[tid] = statsQ[((size_t)b * CNQ + q0) * 2 + tid];
  v8f acc[8];
#pragma unroll
  for (int t = 0; t < 8; ++t) acc[t] = vzero8();

  for (int n0 = 0; n0 < CNN; n0 += 32) {
    __syncthreads();
    if (n0 + 32 < CNN && tid < 32)
      __builtin_prefetch(latF16 + ((size_t)b * CNN + n0 + 32 + tid) * 128, 0, 3);
    for (int i = tid; i < 32 * 128; i += 256) {
      int k = i >> 7, d = i & 127;
      Bt[d * 32 + k] = latF16[((size_t)b * CNN + n0 + k) * 128 + d];
    }
    for (int i = tid; i < 128 * 32; i += 256) {
      int qr = i >> 5, nn = i & 31;
      float p0 = P[(n0 + nn) * 2], p1 = P[(n0 + nn) * 2 + 1];
      float d0 = sq[2 * qr] - p0, d1 = sq[2 * qr + 1] - p1;
      At[i] = (_Float16)(__expf(-(d0 * d0 + d1 * d1) - sst[2 * qr]) * sst[2 * qr + 1]);
    }
    __syncthreads();
    v16h a = load_a_frag(At + (wave * 16) * 32, 32);
#pragma unroll
    for (int t = 0; t < 8; ++t) {
      v16h bf = load_bt_frag(Bt + (t * 16) * 32, 32);
      acc[t] = wmma16(a, bf, acc[t]);
    }
  }
  const int half = lane >> 4, nsub = lane & 15;
#pragma unroll
  for (int t = 0; t < 8; ++t) {
    int col = t * 16 + nsub;
#pragma unroll
    for (int r = 0; r < 8; ++r) {
      int row = wave * 16 + half * 8 + r;
      zf16[((size_t)b * CNQ + q0 + row) * 128 + col] = (_Float16)acc[t][r];
    }
  }
}

// ---------------------------------------------------------------------------
// Edge message MLP: gather [lat[recv] | lat[send]] (K=256) -> 128 -> 128,
// scatter-add into inbox with global f32 atomics. 64 edges / block.
// Gather staging uses async global->LDS DMA (ASYNCcnt) when available.
// ---------------------------------------------------------------------------
__global__ void k_message(const int* __restrict__ recv, const int* __restrict__ send,
                          const _Float16* __restrict__ latF16,
                          const _Float16* __restrict__ wt0, const float* __restrict__ b0,
                          const _Float16* __restrict__ wt1, const float* __restrict__ b1,
                          const _Float16* __restrict__ wt2, const float* __restrict__ b2,
                          float* __restrict__ inbox) {
  __shared__ int sR[64], sS[64];
  __shared__ __align__(16) _Float16 bufA[64 * 256];
  __shared__ __align__(16) _Float16 bufB[64 * 128];
  const int r0  = blockIdx.x * 64;     // flat row in [0, B*E)
  const int bch = r0 / CE;
  const int e0  = r0 % CE;
  if (threadIdx.x < 64) {
    sR[threadIdx.x] = recv[e0 + threadIdx.x];
    sS[threadIdx.x] = send[e0 + threadIdx.x];
  }
  __syncthreads();
  // 16-byte gather chunks: 64 rows x 32 chunks of 8 f16 (async DMA to LDS)
  for (int cchunk = threadIdx.x; cchunk < 64 * 32; cchunk += blockDim.x) {
    int row = cchunk >> 5;
    int seg = cchunk & 31;               // <16 receiver half, >=16 sender half
    int node = (seg < 16) ? sR[row] : sS[row];
    int off  = (seg & 15) * 8;
    async_copy16(latF16 + ((size_t)bch * CNN + node) * 128 + off,
                 bufA + row * 256 + ((seg < 16) ? 0 : 128) + off);
  }
  async_wait0();
  __syncthreads();
  mlp_layer_relu(bufA, 256, bufB, wt0, b0);
  mlp_layer_relu(bufB, 128, bufA, wt1, b1);
  mlp_gemm(bufA, 128, wt2, b2, 8, [&](int r, int c, float v) {
    float* dst = inbox + ((size_t)bch * CNN + sR[r]) * 128 + c;
    __hip_atomic_fetch_add(dst, v, __ATOMIC_RELAXED, __HIP_MEMORY_SCOPE_AGENT);
  });
}

// ---------------------------------------------------------------------------
// Node MLP: [lat | inbox] (K=256) -> 128 -> 128, residual update of latents.
// Latents half staged with async DMA; inbox half converted f32->f16 in VALU.
// ---------------------------------------------------------------------------
__global__ void k_node(float* __restrict__ latF32, _Float16* __restrict__ latF16,
                       const float* __restrict__ inbox,
                       const _Float16* __restrict__ wt0, const float* __restrict__ b0,
                       const _Float16* __restrict__ wt1, const float* __restrict__ b1,
                       const _Float16* __restrict__ wt2, const float* __restrict__ b2) {
  __shared__ __align__(16) _Float16 bufA[64 * 256];
  __shared__ __align__(16) _Float16 bufB[64 * 128];
  const int r0 = blockIdx.x * 64;      // flat row in [0, B*NN)
  for (int cchunk = threadIdx.x; cchunk < 64 * 16; cchunk += blockDim.x) {
    int row = cchunk >> 4;
    int off = (cchunk & 15) * 8;
    async_copy16(latF16 + (size_t)(r0 + row) * 128 + off,
                 bufA + row * 256 + off);
  }
  for (int i = threadIdx.x; i < 64 * 128; i += blockDim.x) {
    int row = i >> 7, c = i & 127;
    bufA[row * 256 + 128 + c] = (_Float16)inbox[(size_t)(r0 + row) * 128 + c];
  }
  async_wait0();
  __syncthreads();
  mlp_layer_relu(bufA, 256, bufB, wt0, b0);
  mlp_layer_relu(bufB, 128, bufA, wt1, b1);
  mlp_gemm(bufA, 128, wt2, b2, 8, [&](int r, int c, float v) {
    size_t off = (size_t)(r0 + r) * 128 + c;
    float nv = latF32[off] + v;
    latF32[off] = nv;
    latF16[off] = (_Float16)nv;
  });
}

// ---------------------------------------------------------------------------
// Decoder: [z | q] (K padded 130->160) -> 128 -> 3.
// ---------------------------------------------------------------------------
__global__ void k_decoder(const _Float16* __restrict__ zf16, const float* __restrict__ qp,
                          const _Float16* __restrict__ wt0, const float* __restrict__ b0,
                          const _Float16* __restrict__ wt1, const float* __restrict__ b1,
                          const _Float16* __restrict__ wt2, const float* __restrict__ b2,
                          float* __restrict__ out) {
  __shared__ __align__(16) _Float16 bufA[64 * 160];
  __shared__ __align__(16) _Float16 bufB[64 * 128];
  const int r0 = blockIdx.x * 64;      // flat row in [0, B*NQ)
  for (int i = threadIdx.x; i < 64 * 160; i += blockDim.x) {
    int row = i / 160, c = i % 160;
    size_t g = (size_t)(r0 + row);
    float v = 0.f;
    if (c < 128)      v = (float)zf16[g * 128 + c];
    else if (c < 130) v = qp[g * 2 + (c - 128)];
    bufA[i] = (_Float16)v;
  }
  __syncthreads();
  mlp_layer_relu(bufA, 160, bufB, wt0, b0);
  mlp_layer_relu(bufB, 128, bufA, wt1, b1);
  mlp_gemm(bufA, 128, wt2, nullptr, 1, [&](int r, int c, float v) {
    if (c < 3) out[(size_t)(r0 + r) * 3 + c] = v + b2[c];
  });
}

__global__ void k_zero(float* __restrict__ p, int n) {
  int i = blockIdx.x * blockDim.x + threadIdx.x;
  if (i < n) p[i] = 0.f;
}

// ---------------------------------------------------------------------------
// Host orchestration
// ---------------------------------------------------------------------------
extern "C" void kernel_launch(void* const* d_in, const int* in_sizes, int n_in,
                              void* d_out, int out_size, void* d_ws, size_t ws_size,
                              hipStream_t stream) {
  (void)in_sizes; (void)n_in; (void)out_size; (void)ws_size;
  // Input order (setup_inputs insertion order, params flattened in source order):
  // 0:x 1:s 2:q 3:senders 4:receivers 5:node_pos
  // 6..11: encoder W0,b0,W1,b1,W2,b2
  // 12+12i .. : block i -> message W0,b0,W1,b1,W2,b2 then node W0,b0,W1,b1,W2,b2
  // 108..113: decoder W0,b0,W1,b1,W2,b2
  const float* x  = (const float*)d_in[0];
  const float* s  = (const float*)d_in[1];
  const float* q  = (const float*)d_in[2];
  const int* senders   = (const int*)d_in[3];
  const int* receivers = (const int*)d_in[4];
  const float* nodePos = (const float*)d_in[5];

  uint8_t* base = (uint8_t*)d_ws;
  size_t cur = 0;
  auto alloc = [&](size_t bytes) -> void* {
    void* p = base + cur;
    cur = (cur + bytes + 255) & ~(size_t)255;
    return p;
  };

  const size_t WTS_ELEMS = 1124352;          // all transposed/padded f16 weights
  _Float16* wts    = (_Float16*)alloc(WTS_ELEMS * 2);
  float*    statsX = (float*)alloc((size_t)CB * CNP * 2 * 4);
  float*    statsQ = (float*)alloc((size_t)CB * CNQ * 2 * 4);
  _Float16* emb    = (_Float16*)alloc((size_t)CB * CNP * 128 * 2);
  float*    latF32 = (float*)alloc((size_t)CB * CNN * 128 * 4);
  _Float16* latF16 = (_Float16*)alloc((size_t)CB * CNN * 128 * 2);
  float*    inbox  = (float*)alloc((size_t)CB * CNN * 128 * 4);
  _Float16* zf16   = (_Float16*)alloc((size_t)CB * CNQ * 128 * 2);

  auto conv = [&](int idx, int r, int c, int kPad, int nPad, size_t off) {
    int total = kPad * nPad;
    k_convert<<<(total + 255) / 256, 256, 0, stream>>>(
        (const float*)d_in[idx], wts + off, r, c, kPad, nPad);
  };

  const size_t OFF_ENC0 = 0, OFF_ENC1 = 4096, OFF_ENC2 = 20480;
  conv(6,  5,   128, 32,  128, OFF_ENC0);
  conv(8,  128, 128, 128, 128, OFF_ENC1);
  conv(10, 128, 128, 128, 128, OFF_ENC2);

  size_t msg0[8], msg1[8], msg2[8], nod0[8], nod1[8], nod2[8];
  for (int i = 0; i < 8; ++i) {
    int bi = 12 + 12 * i;
    size_t o = 36864 + (size_t)131072 * i;
    msg0[i] = o;          msg1[i] = o + 32768;  msg2[i] = o + 49152;
    nod0[i] = o + 65536;  nod1[i] = o + 98304;  nod2[i] = o + 114688;
    conv(bi + 0,  256, 128, 256, 128, msg0[i]);
    conv(bi + 2,  128, 128, 128, 128, msg1[i]);
    conv(bi + 4,  128, 128, 128, 128, msg2[i]);
    conv(bi + 6,  256, 128, 256, 128, nod0[i]);
    conv(bi + 8,  128, 128, 128, 128, nod1[i]);
    conv(bi + 10, 128, 128, 128, 128, nod2[i]);
  }
  const size_t OFF_DEC0 = 1085440, OFF_DEC1 = 1105920, OFF_DEC2 = 1122304;
  conv(108, 130, 128, 160, 128, OFF_DEC0);
  conv(110, 128, 128, 128, 128, OFF_DEC1);
  conv(112, 128, 3,   128, 16,  OFF_DEC2);

  // Softmax stats (pass 1)
  k_stats<<<(CB * CNP) / 256, 256, 0, stream>>>(x, nodePos, statsX, CB * CNP);
  k_stats<<<(CB * CNQ) / 256, 256, 0, stream>>>(q, nodePos, statsQ, CB * CNQ);

  // Encoder
  k_encoder<<<(CB * CNP) / 64, 128, 0, stream>>>(
      x, s, wts + OFF_ENC0, (const float*)d_in[7],
      wts + OFF_ENC1, (const float*)d_in[9],
      wts + OFF_ENC2, (const float*)d_in[11], emb);

  // latents = sc^T @ emb (fused softmax A-tiles)
  k_agg<<<dim3(CNN / 128, CB), 256, 0, stream>>>(x, nodePos, statsX, emb, latF32, latF16);

  // 8 message-passing steps
  for (int i = 0; i < 8; ++i) {
    int bi = 12 + 12 * i;
    k_zero<<<(CB * CNN * 128) / 256, 256, 0, stream>>>(inbox, CB * CNN * 128);
    k_message<<<(CB * CE) / 64, 128, 0, stream>>>(
        receivers, senders, latF16,
        wts + msg0[i], (const float*)d_in[bi + 1],
        wts + msg1[i], (const float*)d_in[bi + 3],
        wts + msg2[i], (const float*)d_in[bi + 5], inbox);
    k_node<<<(CB * CNN) / 64, 128, 0, stream>>>(
        latF32, latF16, inbox,
        wts + nod0[i], (const float*)d_in[bi + 7],
        wts + nod1[i], (const float*)d_in[bi + 9],
        wts + nod2[i], (const float*)d_in[bi + 11]);
  }

  // z = scq @ latents (fused softmax A-tiles)
  k_zq<<<dim3(CNQ / 128, CB), 256, 0, stream>>>(q, nodePos, statsQ, latF16, zf16);

  // Decoder -> d_out
  k_decoder<<<(CB * CNQ) / 64, 128, 0, stream>>>(
      zf16, q, wts + OFF_DEC0, (const float*)d_in[109],
      wts + OFF_DEC1, (const float*)d_in[111],
      wts + OFF_DEC2, (const float*)d_in[113], (float*)d_out);
}
